// NTM_9062380994906
// MI455X (gfx1250) — matrix operand order repaired and checked by reading, not compiled, not run on hardware
//
#include <hip/hip_runtime.h>
#include <hip/hip_bf16.h>

// ---------------------------------------------------------------------------
// NTM forward rollout for MI455X (gfx1250, wave32).
// GEMMs: V_WMMA_F32_16X16X32_BF16, bf16 operands / fp32 accumulate.
// All GEMM B-operands are passed COLUMN-MAJOR (B^T, N x K row-major) so both
// A and B fragments are contiguous 128-bit LDS loads.
// Global->LDS staging is double-buffered and software-pipelined: the next
// K-chunk's global loads are issued before the barrier and consumed a full
// compute phase later, hiding load latency behind the WMMAs.
// ---------------------------------------------------------------------------

typedef __bf16 bf16t;
typedef __attribute__((ext_vector_type(8)))  __bf16 v8bf;
typedef __attribute__((ext_vector_type(16))) __bf16 v16bf;
typedef __attribute__((ext_vector_type(8)))  float  v8f;

#define EPSN 1e-8f

__device__ __forceinline__ bf16t f2bf(float f) {
    unsigned int u = __builtin_bit_cast(unsigned int, f);
    u = (u + 0x7FFFu + ((u >> 16) & 1u)) >> 16;          // round-to-nearest-even
    unsigned short s = (unsigned short)u;
    return __builtin_bit_cast(bf16t, s);
}
__device__ __forceinline__ float sigmoidf_(float x) { return 1.f / (1.f + __expf(-x)); }
__device__ __forceinline__ float softplusf_(float x) { return x > 20.f ? x : log1pf(__expf(x)); }

// ---------------------------------------------------------------------------
// Per-chunk WMMA compute: wave tile 16 x 64 = 4 accumulators.
// As: [64 rows][32 k], Bs: [128 cols][32 k] (transposed tile).
// ---------------------------------------------------------------------------
__device__ __forceinline__ void wmma_chunk(
    const bf16t* __restrict__ As, const bf16t* __restrict__ Bs,
    int lane, int wrow, int wcol, v8f* acc)
{
    // A fragment (16x32): lanes 0-15 row M=lane with K 0..7 & 16..23,
    // lanes 16-31 same rows with K 8..15 & 24..31 (CDNA5 layout).
    const int arow = wrow * 16 + (lane & 15);
    const int kh8  = (lane >> 4) * 8;
    const bf16t* ap = &As[arow * 32 + kh8];
    const v8bf alo = *(const v8bf*)(ap);
    const v8bf ahi = *(const v8bf*)(ap + 16);
    const v16bf afrag = __builtin_shufflevector(
        alo, ahi, 0,1,2,3,4,5,6,7,8,9,10,11,12,13,14,15);

    // B fragments (32x16): lane col = lane%16, K bk0..bk0+7 & bk0+16..bk0+23.
    const int bcolb = wcol * 64 + (lane & 15);
    const int bk0   = (lane >> 4) * 8;
#pragma unroll
    for (int f = 0; f < 4; ++f) {
        const bf16t* bp = &Bs[(bcolb + f * 16) * 32 + bk0];
        const v8bf blo = *(const v8bf*)(bp);
        const v8bf bhi = *(const v8bf*)(bp + 16);
        const v16bf bfrag = __builtin_shufflevector(
            blo, bhi, 0,1,2,3,4,5,6,7,8,9,10,11,12,13,14,15);
        acc[f] = __builtin_amdgcn_wmma_f32_16x16x32_bf16(
            false, afrag, false, bfrag, (short)0, acc[f], false, false);
    }
}

// ---------------------------------------------------------------------------
// bf16 GEMM: C[M,N] = act(A[M,K] @ B[K,N] + bias + (acc ? C : 0))
//   A:  row-major M x K (lda);  BT: row-major N x K (ldbT)
// Block = 256 threads = 8 waves; block tile 64 x 128; wave tile 16 x 64;
// K in chunks of 32, double-buffered through LDS with register pipelining.
// Requires: M%64==0, N%128==0, K%32==0, lda/ldbT multiples of 8.
// ---------------------------------------------------------------------------
__global__ __launch_bounds__(256) void gemm_bf16_kernel(
    const bf16t* __restrict__ A, int lda,
    const bf16t* __restrict__ BT, int ldbT,
    const float* __restrict__ bias, int nbias,
    float* __restrict__ C, int ldc,
    int K, int act, int accumulate,
    bf16t* __restrict__ Cbf, int ldcbf)
{
    __shared__ __align__(16) bf16t As[2][64 * 32];
    __shared__ __align__(16) bf16t Bs[2][128 * 32];
    const int tid  = threadIdx.x;
    const int lane = tid & 31;
    const int wave = tid >> 5;
    const int wrow = wave >> 1;            // 0..3 : 16-row strip
    const int wcol = wave & 1;             // 0..1 : 64-col strip
    const long row0 = (long)blockIdx.y * 64;
    const long col0 = (long)blockIdx.x * 128;

    v8f acc[4];
#pragma unroll
    for (int f = 0; f < 4; ++f) acc[f] = (v8f){0.f,0.f,0.f,0.f,0.f,0.f,0.f,0.f};

    const int ar = tid >> 2, ac = (tid & 3) * 8;     // A: 64 rows x 32 k
    const int bn = tid >> 1, bk = (tid & 1) * 16;    // B: 128 cols x 32 k
    const bf16t* aSrc = &A[(row0 + ar) * (long)lda + ac];
    const bf16t* bSrc = &BT[(col0 + bn) * (long)ldbT + bk];
    bf16t* aDst0 = &As[0][ar * 32 + ac];
    bf16t* bDst0 = &Bs[0][bn * 32 + bk];
    bf16t* aDst1 = &As[1][ar * 32 + ac];
    bf16t* bDst1 = &Bs[1][bn * 32 + bk];
    const int nChunks = K >> 5;

    // Register-pipelined staging: chunk i+1's global loads are in flight while
    // chunk i's WMMAs execute; the loadcnt wait lands a full phase after issue.
    uint4 aR  = *(const uint4*)(aSrc);
    uint4 bR0 = *(const uint4*)(bSrc);
    uint4 bR1 = *(const uint4*)(bSrc + 8);
    for (int i = 0; i < nChunks; ++i) {
        const int cur = i & 1;
        *(uint4*)(cur ? aDst1 : aDst0)       = aR;
        *(uint4*)(cur ? bDst1 : bDst0)       = bR0;
        *(uint4*)((cur ? bDst1 : bDst0) + 8) = bR1;
        if (i + 1 < nChunks) {
            const bf16t* a2 = aSrc + (long)(i + 1) * 32;
            const bf16t* b2 = bSrc + (long)(i + 1) * 32;
            aR  = *(const uint4*)(a2);
            bR0 = *(const uint4*)(b2);
            bR1 = *(const uint4*)(b2 + 8);
        }
        __syncthreads();
        wmma_chunk(As[cur], Bs[cur], lane, wrow, wcol, acc);
    }

    // D layout: VGPR i -> row i (lanes 0-15) / i+8 (lanes 16-31), col = lane%16
    const int scol = lane & 15;
    const int srow = (lane >> 4) * 8;
#pragma unroll
    for (int f = 0; f < 4; ++f) {
        const long gc = col0 + wcol * 64 + f * 16 + scol;
        const float bv = (bias && gc < nbias) ? bias[gc] : 0.f;
#pragma unroll
        for (int i = 0; i < 8; ++i) {
            const long r = row0 + wrow * 16 + srow + i;
            float v = acc[f][i] + bv;
            if (accumulate) v += C[r * (long)ldc + gc];
            if (act == 1) v = tanhf(v);
            else if (act == 2) v = sigmoidf_(v);
            C[r * (long)ldc + gc] = v;
            if (Cbf) Cbf[r * (long)ldcbf + gc] = f2bf(v);
        }
    }
}

// ---------------------------------------------------------------------------
// Block reductions (256 threads)
// ---------------------------------------------------------------------------
__device__ __forceinline__ float blk_sum(float v, float* red) {
    const int tid = threadIdx.x;
    red[tid] = v; __syncthreads();
    for (int s = 128; s > 0; s >>= 1) {
        if (tid < s) red[tid] += red[tid + s];
        __syncthreads();
    }
    float r = red[0]; __syncthreads();
    return r;
}
__device__ __forceinline__ float blk_max(float v, float* red) {
    const int tid = threadIdx.x;
    red[tid] = v; __syncthreads();
    for (int s = 128; s > 0; s >>= 1) {
        if (tid < s) red[tid] = fmaxf(red[tid], red[tid + s]);
        __syncthreads();
    }
    float r = red[0]; __syncthreads();
    return r;
}

// ---------------------------------------------------------------------------
// Conversion / elementwise kernels
// ---------------------------------------------------------------------------
__global__ __launch_bounds__(256) void f32_to_bf16_kernel(
    const float* __restrict__ s, bf16t* __restrict__ d, int n)
{
    const int i = blockIdx.x * 256 + threadIdx.x;
    if (i < n) d[i] = f2bf(s[i]);
}

// dst[c*rows + r] = src[r*cols + c]  (produces B^T for row-major weights)
__global__ __launch_bounds__(256) void transpose_f32_to_bf16_kernel(
    const float* __restrict__ s, bf16t* __restrict__ d, int rows, int cols)
{
    const int i = blockIdx.x * 256 + threadIdx.x;
    if (i >= rows * cols) return;
    const int r = i / cols, c = i - r * cols;
    d[c * rows + r] = f2bf(s[i]);
}

// dst (dcols x rows): dst[n*rows + h] = n < scols ? src[h*scols + n] : 0
__global__ __launch_bounds__(256) void transpose_pad_f32_to_bf16_kernel(
    const float* __restrict__ s, bf16t* __restrict__ d,
    int rows, int scols, int dcols)
{
    const int i = blockIdx.x * 256 + threadIdx.x;
    if (i >= rows * dcols) return;
    const int n = i / rows, h = i - n * rows;
    d[i] = (n < scols) ? f2bf(s[h * scols + n]) : f2bf(0.f);
}

__global__ __launch_bounds__(256) void copy_f32_kernel(
    const float* __restrict__ s, float* __restrict__ d, int n)
{
    const int i = blockIdx.x * 256 + threadIdx.x;
    if (i < n) d[i] = s[i];
}

// ea (B x 1024) -> eT[m*256+b] = sigmoid(ea[b,m]), aT[m*256+b] = tanh(ea[b,512+m])
__global__ __launch_bounds__(256) void ea_split_kernel(
    const float* __restrict__ ea, bf16t* __restrict__ eT, bf16t* __restrict__ aT)
{
    const int i = blockIdx.x * 256 + threadIdx.x;   // over B*512
    const int b = i >> 9, m = i & 511;
    eT[m * 256 + b] = f2bf(sigmoidf_(ea[b * 1024 + m]));
    aT[m * 256 + b] = f2bf(tanhf(ea[b * 1024 + 512 + m]));
}

// mem = mem*(1 - we/B) + wa/B ; bf16 shadow stored transposed (M x N)
__global__ __launch_bounds__(256) void mem_update_kernel(
    float* __restrict__ mem, const float* __restrict__ we,
    const float* __restrict__ wa, bf16t* __restrict__ memT_bf, int n)
{
    const int i = blockIdx.x * 256 + threadIdx.x;
    if (i >= n) return;
    const int nn = i >> 9, m = i & 511;
    const float inv_b = 1.f / 256.f;
    const float v = mem[i] * (1.f - we[i] * inv_b) + wa[i] * inv_b;
    mem[i] = v;
    memT_bf[m * 1024 + nn] = f2bf(v);
}

// per memory row n: mn_bf[n, :] = mem[n, :] / (||mem[n]|| + eps)
// (row-major N x M == transposed form of the content-GEMM B operand mn^T)
__global__ __launch_bounds__(256) void mem_norm_kernel(
    const float* __restrict__ mem, bf16t* __restrict__ mn_bf)
{
    __shared__ float red[256];
    const int n = blockIdx.x, tid = threadIdx.x;
    const float m0 = mem[n * 512 + tid];
    const float m1 = mem[n * 512 + 256 + tid];
    const float ss = blk_sum(m0 * m0 + m1 * m1, red);
    const float inv = 1.f / (sqrtf(ss) + EPSN);
    mn_bf[n * 512 + tid]       = f2bf(m0 * inv);
    mn_bf[n * 512 + 256 + tid] = f2bf(m1 * inv);
}

// head params: p (B x 640, cols 0..517 valid) ->
//   kn (bf16 normalized key, row-major), scal = {beta, g, s0, s1, s2, gamma}
__global__ __launch_bounds__(256) void address_prep_kernel(
    const float* __restrict__ p, bf16t* __restrict__ kn, float* __restrict__ scal)
{
    __shared__ float red[256];
    const int b = blockIdx.x, tid = threadIdx.x;
    const float* pr = p + b * 640;
    const float k0 = tanhf(pr[tid]);
    const float k1 = tanhf(pr[256 + tid]);
    const float ss = blk_sum(k0 * k0 + k1 * k1, red);
    const float inv = 1.f / (sqrtf(ss) + EPSN);
    kn[b * 512 + tid]       = f2bf(k0 * inv);
    kn[b * 512 + 256 + tid] = f2bf(k1 * inv);
    if (tid == 0) {
        const float beta = softplusf_(pr[512]);
        const float g    = sigmoidf_(pr[513]);
        const float a0 = pr[514], a1 = pr[515], a2 = pr[516];
        const float mx = fmaxf(a0, fmaxf(a1, a2));
        const float e0 = __expf(a0 - mx), e1 = __expf(a1 - mx), e2 = __expf(a2 - mx);
        const float si = 1.f / (e0 + e1 + e2);
        scal[b * 8 + 0] = beta;
        scal[b * 8 + 1] = g;
        scal[b * 8 + 2] = e0 * si;
        scal[b * 8 + 3] = e1 * si;
        scal[b * 8 + 4] = e2 * si;
        scal[b * 8 + 5] = 1.f + softplusf_(pr[517]);
    }
}

// softmax(beta*sim) -> interpolate -> circular shift -> sharpen -> normalize
// w (fp32, in/out row b). wbf: bf16 copy (transposed N x B if transposed=1).
__global__ __launch_bounds__(256) void address_finish_kernel(
    const float* __restrict__ wcraw, const float* __restrict__ scal,
    float* __restrict__ w, bf16t* __restrict__ wbf, int transposed)
{
    __shared__ float wg[1024];
    __shared__ float red[256];
    const int b = blockIdx.x, tid = threadIdx.x;
    const float beta = scal[b * 8 + 0], g = scal[b * 8 + 1];
    const float s0 = scal[b * 8 + 2], s1 = scal[b * 8 + 3], s2 = scal[b * 8 + 4];
    const float gamma = scal[b * 8 + 5];

    float v[4];
    float mx = -3.4e38f;
#pragma unroll
    for (int j = 0; j < 4; ++j) {
        v[j] = beta * wcraw[b * 1024 + tid + j * 256];
        mx = fmaxf(mx, v[j]);
    }
    mx = blk_max(mx, red);
    float se = 0.f;
#pragma unroll
    for (int j = 0; j < 4; ++j) { v[j] = __expf(v[j] - mx); se += v[j]; }
    se = blk_sum(se, red);
    const float invsm = 1.f / se;
#pragma unroll
    for (int j = 0; j < 4; ++j) {
        const int i = tid + j * 256;
        wg[i] = g * (v[j] * invsm) + (1.f - g) * w[b * 1024 + i];
    }
    __syncthreads();
    float ssum = 0.f;
#pragma unroll
    for (int j = 0; j < 4; ++j) {
        const int i = tid + j * 256;
        const float wt = s0 * wg[(i + 1) & 1023] + s1 * wg[i] + s2 * wg[(i + 1023) & 1023];
        v[j] = __powf(wt, gamma);
        ssum += v[j];
    }
    ssum = blk_sum(ssum, red);
    const float invn = 1.f / (ssum + EPSN);
#pragma unroll
    for (int j = 0; j < 4; ++j) {
        const int i = tid + j * 256;
        const float wn = v[j] * invn;
        w[b * 1024 + i] = wn;
        if (transposed) wbf[i * 256 + b] = f2bf(wn);
        else            wbf[b * 1024 + i] = f2bf(wn);
    }
}

// ---------------------------------------------------------------------------
// Host driver
// ---------------------------------------------------------------------------
extern "C" void kernel_launch(void* const* d_in, const int* in_sizes, int n_in,
                              void* d_out, int out_size, void* d_ws, size_t ws_size,
                              hipStream_t stream)
{
    const int Bn = 256, T = 128, In = 512, Hn = 512, Nn = 1024, Mn = 512, On = 512;
    const int PH = 640;  // padded head-param width (518 -> 640 = 5*128)

    const float* x    = (const float*)d_in[0];
    const float* mem0 = (const float*)d_in[1];
    const float* wr0  = (const float*)d_in[2];
    const float* ww0  = (const float*)d_in[3];
    const float* h0   = (const float*)d_in[4];
    const float* Wx   = (const float*)d_in[5];
    const float* Wr   = (const float*)d_in[6];
    const float* bh   = (const float*)d_in[7];
    const float* Whr  = (const float*)d_in[8];
    const float* bhr  = (const float*)d_in[9];
    const float* Whw  = (const float*)d_in[10];
    const float* bhw  = (const float*)d_in[11];
    const float* Wea  = (const float*)d_in[12];
    const float* bea  = (const float*)d_in[13];
    const float* Wo   = (const float*)d_in[14];
    const float* bo   = (const float*)d_in[15];
    float* out = (float*)d_out;

    char* wsp = (char*)d_ws;
    auto alloc = [&](size_t elems, size_t esz) -> void* {
        void* r = (void*)wsp;
        wsp += (elems * esz + 255) & ~(size_t)255;
        return r;
    };

    // bf16 constants (weights stored transposed: N x K row-major)
    bf16t* x_bf  = (bf16t*)alloc((size_t)Bn * T * In, 2);
    bf16t* WxT   = (bf16t*)alloc((size_t)Hn * In, 2);
    bf16t* WrT   = (bf16t*)alloc((size_t)Hn * Mn, 2);
    bf16t* WeaT  = (bf16t*)alloc((size_t)2 * Mn * Hn, 2);
    bf16t* WoT   = (bf16t*)alloc((size_t)On * Hn, 2);
    bf16t* WhrT  = (bf16t*)alloc((size_t)PH * Hn, 2);
    bf16t* WhwT  = (bf16t*)alloc((size_t)PH * Hn, 2);
    // fp32 state
    float* mem = (float*)alloc((size_t)Nn * Mn, 4);
    float* wr  = (float*)alloc((size_t)Bn * Nn, 4);
    float* ww  = (float*)alloc((size_t)Bn * Nn, 4);
    // bf16 state shadows
    bf16t* h_bf    = (bf16t*)alloc((size_t)Bn * Hn, 2);
    bf16t* r_bf    = (bf16t*)alloc((size_t)Bn * Mn, 2);
    bf16t* eT_bf   = (bf16t*)alloc((size_t)Mn * Bn, 2);
    bf16t* aT_bf   = (bf16t*)alloc((size_t)Mn * Bn, 2);
    bf16t* kn_bf   = (bf16t*)alloc((size_t)Bn * Mn, 2);
    bf16t* wr_bf   = (bf16t*)alloc((size_t)Bn * Nn, 2);
    bf16t* wwT_bf  = (bf16t*)alloc((size_t)Nn * Bn, 2);
    bf16t* memT_bf = (bf16t*)alloc((size_t)Mn * Nn, 2);
    bf16t* mn_bf   = (bf16t*)alloc((size_t)Nn * Mn, 2);
    // fp32 scratch
    float* ea    = (float*)alloc((size_t)Bn * 2 * Mn, 4);
    float* we    = (float*)alloc((size_t)Nn * Mn, 4);
    float* wa    = (float*)alloc((size_t)Nn * Mn, 4);
    float* hpre  = (float*)alloc((size_t)Bn * Hn, 4);
    float* rtmp  = (float*)alloc((size_t)Bn * Mn, 4);
    float* pbuf  = (float*)alloc((size_t)Bn * PH, 4);
    float* wcraw = (float*)alloc((size_t)Bn * Nn, 4);
    float* scal  = (float*)alloc((size_t)Bn * 8, 4);

    auto cvt = [&](const float* s, bf16t* d, int n) {
        f32_to_bf16_kernel<<<(n + 255) / 256, 256, 0, stream>>>(s, d, n);
    };
    auto tra = [&](const float* s, bf16t* d, int rows, int cols) {
        transpose_f32_to_bf16_kernel<<<(rows * cols + 255) / 256, 256, 0, stream>>>(s, d, rows, cols);
    };
    auto gemm = [&](const bf16t* A, int lda, const bf16t* BT, int ldbT,
                    const float* bias, int nbias, float* C, int ldc,
                    int Mr, int Nc, int K, int act, int acc,
                    bf16t* Cbf, int ldcbf) {
        dim3 g(Nc / 128, Mr / 64);
        gemm_bf16_kernel<<<g, 256, 0, stream>>>(A, lda, BT, ldbT, bias, nbias,
                                                C, ldc, K, act, acc, Cbf, ldcbf);
    };

    // ---- setup: bf16 conversions + state init -----------------------------
    cvt(x, x_bf, Bn * T * In);
    tra(Wx, WxT, In, Hn);                 // WxT: Hn x In
    tra(Wr, WrT, Mn, Hn);                 // WrT: Hn x Mn
    tra(Wea, WeaT, Hn, 2 * Mn);           // WeaT: 2Mn x Hn
    tra(Wo, WoT, Hn, On);                 // WoT: On x Hn
    transpose_pad_f32_to_bf16_kernel<<<(PH * Hn + 255) / 256, 256, 0, stream>>>(Whr, WhrT, Hn, Mn + 6, PH);
    transpose_pad_f32_to_bf16_kernel<<<(PH * Hn + 255) / 256, 256, 0, stream>>>(Whw, WhwT, Hn, Mn + 6, PH);
    cvt(h0, h_bf, Bn * Hn);
    cvt(wr0, wr_bf, Bn * Nn);
    transpose_f32_to_bf16_kernel<<<(Bn * Nn + 255) / 256, 256, 0, stream>>>(ww0, wwT_bf, Bn, Nn);
    copy_f32_kernel<<<(Nn * Mn + 255) / 256, 256, 0, stream>>>(mem0, mem, Nn * Mn);
    copy_f32_kernel<<<(Bn * Nn + 255) / 256, 256, 0, stream>>>(wr0, wr, Bn * Nn);
    copy_f32_kernel<<<(Bn * Nn + 255) / 256, 256, 0, stream>>>(ww0, ww, Bn * Nn);

    // ---- sequential rollout ----------------------------------------------
    for (int t = 0; t < T; ++t) {
        // 1) write head erase/add: ea = h @ Wea + bea
        gemm(h_bf, Hn, WeaT, Hn, bea, 2 * Mn, ea, 2 * Mn,
             Bn, 2 * Mn, Hn, 0, 0, nullptr, 0);
        ea_split_kernel<<<(Bn * Mn) / 256, 256, 0, stream>>>(ea, eT_bf, aT_bf);

        // 2) memory update: we = ww^T@e, wa = ww^T@a, then blend
        gemm(wwT_bf, Bn, eT_bf, Bn, nullptr, 0, we, Mn, Nn, Mn, Bn, 0, 0, nullptr, 0);
        gemm(wwT_bf, Bn, aT_bf, Bn, nullptr, 0, wa, Mn, Nn, Mn, Bn, 0, 0, nullptr, 0);
        mem_update_kernel<<<(Nn * Mn) / 256, 256, 0, stream>>>(mem, we, wa, memT_bf, Nn * Mn);
        mem_norm_kernel<<<Nn, 256, 0, stream>>>(mem, mn_bf);

        // 3) read: r = wr @ mem   (B operand = memT_bf, col-major mem)
        gemm(wr_bf, Nn, memT_bf, Nn, nullptr, 0, rtmp, Mn, Bn, Mn, Nn, 0, 0, r_bf, Mn);

        // 4) controller: h = tanh(x_t @ Wx + r @ Wr + bh)
        gemm(x_bf + (size_t)t * In, T * In, WxT, In, bh, Hn, hpre, Hn,
             Bn, Hn, In, 0, 0, nullptr, 0);
        gemm(r_bf, Mn, WrT, Mn, nullptr, 0, hpre, Hn, Bn, Hn, Mn, 1, 1, h_bf, Hn);

        // 5) write-head addressing (updates ww, wwT_bf)
        gemm(h_bf, Hn, WhwT, Hn, bhw, Mn + 6, pbuf, PH, Bn, PH, Hn, 0, 0, nullptr, 0);
        address_prep_kernel<<<Bn, 256, 0, stream>>>(pbuf, kn_bf, scal);
        gemm(kn_bf, Mn, mn_bf, Mn, nullptr, 0, wcraw, Nn, Bn, Nn, Mn, 0, 0, nullptr, 0);
        address_finish_kernel<<<Bn, 256, 0, stream>>>(wcraw, scal, ww, wwT_bf, 1);

        // 6) read-head addressing (updates wr, wr_bf)
        gemm(h_bf, Hn, WhrT, Hn, bhr, Mn + 6, pbuf, PH, Bn, PH, Hn, 0, 0, nullptr, 0);
        address_prep_kernel<<<Bn, 256, 0, stream>>>(pbuf, kn_bf, scal);
        gemm(kn_bf, Mn, mn_bf, Mn, nullptr, 0, wcraw, Nn, Bn, Nn, Mn, 0, 0, nullptr, 0);
        address_finish_kernel<<<Bn, 256, 0, stream>>>(wcraw, scal, wr, wr_bf, 0);

        // 7) output: out[:, t, :] = sigmoid(h @ Wo + bo)
        gemm(h_bf, Hn, WoT, Hn, bo, On, out + (size_t)t * On, T * On,
             Bn, On, Hn, 2, 0, nullptr, 0);
    }
}
